// GPT_17695265259906
// MI455X (gfx1250) — matrix operand order repaired and checked by reading, not compiled, not run on hardware
//
#include <hip/hip_runtime.h>

// ---------------------------------------------------------------------------
// Decision-Transformer forward for MI455X (gfx1250, wave32, WMMA).
// All big GEMMs + attention matmuls use v_wmma_f32_16x16x32_f16.
// GEMM: 128x128x32 macro-tile, 8 waves x (32x64) sub-tiles = 64 WMMA / K-step,
// float4-vectorized weight loads with f32->f16 transpose into LDS, and
// global_prefetch of the next K-tile.
// ---------------------------------------------------------------------------

typedef __attribute__((ext_vector_type(16))) _Float16 v16h;
typedef __attribute__((ext_vector_type(8)))  _Float16 v8h;
typedef __attribute__((ext_vector_type(8)))  float    v8f;
typedef __attribute__((ext_vector_type(4)))  float    v4f;

#define NB   8
#define NT   512
#define NS   1024          // 2*T
#define ND   768
#define NH   12
#define NDH  64
#define NL   12
#define NDFF 3072
#define NOBS 128
#define MTOK (NB * NS)     // 8192 token rows

// ---------------- WMMA fragment loaders (contiguous k = 16*hl + s, not the ISA table order; A and B agree) ----------

// A 16x32 f16: lanes 0-15 row=r, elems 0..15 -> K 0..15;
// lanes 16-31 same rows, K 16..31.  p points at (row, K=0); hl=lane>>4.
static __device__ inline v16h afrag16x32(const _Float16* p, int hl) {
  const _Float16* q = p + hl * 16;
  v8h lo = *(const v8h*)q;
  v8h hi = *(const v8h*)(q + 8);
  v16h r;
#pragma unroll
  for (int j = 0; j < 8; ++j) { r[j] = lo[j]; r[j + 8] = hi[j]; }
  return r;
}

// B 32x16 f16: lane n holds column n; lanes 0-15 K=0..15, lanes 16-31 K=16..31.
// p points at 16 contiguous K values for this lane's column.
static __device__ inline v16h bfrag32x16(const _Float16* p) {
  v8h lo = *(const v8h*)p;
  v8h hi = *(const v8h*)(p + 8);
  v16h r;
#pragma unroll
  for (int j = 0; j < 8; ++j) { r[j] = lo[j]; r[j + 8] = hi[j]; }
  return r;
}

static __device__ inline v8f wmma_f16(v16h a, v16h b, v8f c) {
  return __builtin_amdgcn_wmma_f32_16x16x32_f16(false, a, false, b,
                                                (short)0, c, false, false);
}

// ---------------- Embedding / interleave / positional -----------------------

__global__ __launch_bounds__(256) void embed_kernel(
    const float* __restrict__ states, const int* __restrict__ actions,
    const int* __restrict__ timesteps, const float* __restrict__ W_s,
    const float* __restrict__ b_s, const float* __restrict__ A_emb,
    const float* __restrict__ pos_emb, const float* __restrict__ gpos_emb,
    float* __restrict__ X) {
  __shared__ float st[NOBS];
  int blk = blockIdx.x;            // b*T + t
  int b = blk >> 9;                // T = 512
  int t = blk & 511;
  int tid = threadIdx.x;
  if (tid < NOBS) st[tid] = states[(size_t)blk * NOBS + tid];
  __syncthreads();
  int act = actions[blk];
  int ts  = timesteps[b];
  int s0 = 2 * t, s1 = 2 * t + 1;
  for (int d = tid; d < ND; d += 256) {
    float acc = b_s[d];
#pragma unroll 8
    for (int o = 0; o < NOBS; ++o) acc += st[o] * W_s[(size_t)o * ND + d];
    float se = tanhf(acc);
    float ae = tanhf(A_emb[(size_t)act * ND + d]);
    float gp = gpos_emb[(size_t)ts * ND + d];
    X[((size_t)b * NS + s0) * ND + d] = se + gp + pos_emb[(size_t)s0 * ND + d];
    X[((size_t)b * NS + s1) * ND + d] = ae + gp + pos_emb[(size_t)s1 * ND + d];
  }
}

// ---------------- LayerNorm: f32 row -> f16 row ------------------------------

__global__ __launch_bounds__(256) void ln_f16_kernel(
    const float* __restrict__ X, const float* __restrict__ g,
    const float* __restrict__ bb, _Float16* __restrict__ O) {
  __shared__ float red[256];
  int row = blockIdx.x, tid = threadIdx.x;
  const float* xr = X + (size_t)row * ND;
  float v0 = xr[tid], v1 = xr[tid + 256], v2 = xr[tid + 512];
  red[tid] = v0 + v1 + v2;
  __syncthreads();
  for (int st = 128; st > 0; st >>= 1) {
    if (tid < st) red[tid] += red[tid + st];
    __syncthreads();
  }
  float mean = red[0] * (1.f / ND);
  __syncthreads();
  float d0 = v0 - mean, d1 = v1 - mean, d2 = v2 - mean;
  red[tid] = d0 * d0 + d1 * d1 + d2 * d2;
  __syncthreads();
  for (int st = 128; st > 0; st >>= 1) {
    if (tid < st) red[tid] += red[tid + st];
    __syncthreads();
  }
  float inv = rsqrtf(red[0] * (1.f / ND) + 1e-5f);
  O[(size_t)row * ND + tid]       = (_Float16)(d0 * inv * g[tid]       + bb[tid]);
  O[(size_t)row * ND + tid + 256] = (_Float16)(d1 * inv * g[tid + 256] + bb[tid + 256]);
  O[(size_t)row * ND + tid + 512] = (_Float16)(d2 * inv * g[tid + 512] + bb[tid + 512]);
}

// Final LN: reads even interleaved rows of x, writes f32 output [B,T,D].
__global__ __launch_bounds__(256) void lnf_kernel(
    const float* __restrict__ X, const float* __restrict__ g,
    const float* __restrict__ bb, float* __restrict__ O) {
  __shared__ float red[256];
  int blk = blockIdx.x;            // b*T + t
  int b = blk >> 9, t = blk & 511;
  int srow = b * NS + 2 * t;
  int tid = threadIdx.x;
  const float* xr = X + (size_t)srow * ND;
  float v0 = xr[tid], v1 = xr[tid + 256], v2 = xr[tid + 512];
  red[tid] = v0 + v1 + v2;
  __syncthreads();
  for (int st = 128; st > 0; st >>= 1) {
    if (tid < st) red[tid] += red[tid + st];
    __syncthreads();
  }
  float mean = red[0] * (1.f / ND);
  __syncthreads();
  float d0 = v0 - mean, d1 = v1 - mean, d2 = v2 - mean;
  red[tid] = d0 * d0 + d1 * d1 + d2 * d2;
  __syncthreads();
  for (int st = 128; st > 0; st >>= 1) {
    if (tid < st) red[tid] += red[tid + st];
    __syncthreads();
  }
  float inv = rsqrtf(red[0] * (1.f / ND) + 1e-5f);
  O[(size_t)blk * ND + tid]       = d0 * inv * g[tid]       + bb[tid];
  O[(size_t)blk * ND + tid + 256] = d1 * inv * g[tid + 256] + bb[tid + 256];
  O[(size_t)blk * ND + tid + 512] = d2 * inv * g[tid + 512] + bb[tid + 512];
}

// ---------------- WMMA GEMM: out = act(A_f16[M,K] @ W_f32[K,N] + bias) ------
// ACT: 0 none, 1 exact GELU.
// STORE: 0 = f16 plain [M,N]; 1 = f16 scattered to [B,H,S,DH]; 2 = f32 x += .

#define BM 128
#define BN 128
#define BK 32
#define LDA_P (BK + 8)   // padded LDS strides (halfs); *2B stays 16B-aligned
#define LDB_P (BK + 8)

template <int ACT, int STORE>
__global__ __launch_bounds__(256) void gemm_kernel(
    const _Float16* __restrict__ A, const float* __restrict__ W,
    const float* __restrict__ bias, float* __restrict__ X,
    _Float16* __restrict__ O16, int M, int N, int K) {
  __shared__ _Float16 sA[BM * LDA_P];
  __shared__ _Float16 sB[BN * LDB_P];
  int tid = threadIdx.x;
  int lane = tid & 31, wv = tid >> 5;     // 8 waves
  int wr = wv >> 1, wc = wv & 1;          // 4x2 wave grid, 32x64 per wave
  int m0 = blockIdx.y * BM;
  int n0 = blockIdx.x * BN;
  int hl = lane >> 4, r16 = lane & 15;

  v8f acc[2][4] = {};
  for (int k0 = 0; k0 < K; k0 += BK) {
    // A tile 128x32 f16: two 8-half chunks per thread, coalesced b128 loads.
#pragma unroll
    for (int c = 0; c < 2; ++c) {
      int ch  = tid + c * 256;            // 0..511 chunk id
      int row = ch >> 2;                  // 128 rows
      int cc  = (ch & 3) * 8;             // 4 chunks of 8 halfs per row
      *(v8h*)(sA + row * LDA_P + cc) =
          *(const v8h*)(A + (size_t)(m0 + row) * K + k0 + cc);
    }
    // W tile 32x128 f32, float4-vectorized, f16-converted, transposed sB[n][k].
#pragma unroll
    for (int c = 0; c < 4; ++c) {
      int f  = tid + c * 256;             // 0..1023 float4 id
      int kk = f >> 5;                    // 32 float4s per k-row
      int nn = (f & 31) * 4;
      v4f w = *(const v4f*)(W + (size_t)(k0 + kk) * N + n0 + nn);
#pragma unroll
      for (int q = 0; q < 4; ++q) sB[(nn + q) * LDB_P + kk] = (_Float16)w[q];
    }
    // Prefetch next K-tile of the weight stream + activation stream into L2.
    if (k0 + BK < K) {
      __builtin_prefetch(W + (size_t)(k0 + BK + (tid >> 3)) * N + n0 + (tid & 7) * 16, 0, 1);
      __builtin_prefetch(A + (size_t)(m0 + (tid >> 1)) * K + k0 + BK + (tid & 1) * 16, 0, 1);
    }
    __syncthreads();
    v16h af[2], bf[4];
#pragma unroll
    for (int i = 0; i < 2; ++i)
      af[i] = afrag16x32(sA + (wr * 32 + i * 16 + r16) * LDA_P, hl);
#pragma unroll
    for (int j = 0; j < 4; ++j)
      bf[j] = bfrag32x16(sB + (wc * 64 + j * 16 + r16) * LDB_P + hl * 16);
#pragma unroll
    for (int i = 0; i < 2; ++i)
#pragma unroll
      for (int j = 0; j < 4; ++j) acc[i][j] = wmma_f16(af[i], bf[j], acc[i][j]);
    __syncthreads();
  }

  int rbase = m0 + wr * 32, cbase = n0 + wc * 64;
#pragma unroll
  for (int i = 0; i < 2; ++i)
#pragma unroll
    for (int j = 0; j < 4; ++j) {
      int col = cbase + j * 16 + r16;
      float bv = bias[col];
#pragma unroll
      for (int v = 0; v < 8; ++v) {
        int row = rbase + i * 16 + v + 8 * hl;   // C layout: VGPR v -> rows v / v+8
        float val = acc[i][j][v] + bv;
        if (ACT == 1) val = 0.5f * val * (1.0f + erff(val * 0.70710678118654752f));
        if (STORE == 0) {
          O16[(size_t)row * N + col] = (_Float16)val;
        } else if (STORE == 1) {
          int bi = row >> 10, si = row & 1023;   // S = 1024
          int hh = col >> 6,  dh = col & 63;     // DH = 64
          O16[((((size_t)(bi * NH + hh)) << 10) + si) * NDH + dh] = (_Float16)val;
        } else {
          X[(size_t)row * N + col] += val;
        }
      }
    }
}

// ---------------- Flash attention (WMMA for QK^T and P@V) --------------------
// One wave per (b,h, 16-query tile); 32-key chunks; online softmax.

#define ACK 32               // keys per chunk
#define ANQ 16               // queries per wave
#define LQ_P (NDH + 8)       // 72
#define LK_P (NDH + 8)       // 72
#define LV_P (ACK + 8)       // 40
#define LP_P (ACK + 8)       // 40
#define AW_HALFS (ANQ * LQ_P + ACK * LK_P + NDH * LV_P + ANQ * LP_P)
#define AWAVES 4

__global__ __launch_bounds__(128) void attn_kernel(
    const _Float16* __restrict__ Q, const _Float16* __restrict__ Kt,
    const _Float16* __restrict__ V, _Float16* __restrict__ Y) {
  __shared__ _Float16 smem[AWAVES * AW_HALFS];
  int tid = threadIdx.x, lane = tid & 31, wv = tid >> 5;
  _Float16* sQ = smem + wv * AW_HALFS;
  _Float16* sK = sQ + ANQ * LQ_P;
  _Float16* sV = sK + ACK * LK_P;
  _Float16* sP = sV + NDH * LV_P;

  int gw = blockIdx.x * AWAVES + wv;      // global wave id
  int qt = gw & 63;                       // S/16 = 64 query tiles
  int bh = gw >> 6;                       // 0..95
  int q0 = qt * ANQ;
  int b = bh / NH, h = bh % NH;
  const _Float16* Qb = Q  + (size_t)bh * NS * NDH;
  const _Float16* Kb = Kt + (size_t)bh * NS * NDH;
  const _Float16* Vb = V  + (size_t)bh * NS * NDH;
  int hl = lane >> 4, r16 = lane & 15;

  // stage Q tile (16x64) into LDS
  for (int e = lane; e < ANQ * NDH; e += 32) {
    int r = e >> 6, c = e & 63;
    sQ[r * LQ_P + c] = Qb[(size_t)(q0 + r) * NDH + c];
  }
  __builtin_amdgcn_wave_barrier();
  asm volatile("s_wait_dscnt 0" ::: "memory");

  float mrow[8], lrow[8];
  v8f o[4] = {};
#pragma unroll
  for (int i = 0; i < 8; ++i) { mrow[i] = -1e30f; lrow[i] = 0.f; }

  const float sc = 0.125f;                 // 1/sqrt(DH)
  int kmax = q0 + ANQ;
  for (int kc = 0; kc < kmax; kc += ACK) {
    // stage K chunk (32x64) and V chunk transposed (64x32)
    for (int e = lane; e < ACK * NDH; e += 32) {
      int r = e >> 6, c = e & 63;
      _Float16 kvK = Kb[(size_t)(kc + r) * NDH + c];
      _Float16 kvV = Vb[(size_t)(kc + r) * NDH + c];
      sK[r * LK_P + c] = kvK;
      sV[c * LV_P + r] = kvV;
    }
    __builtin_amdgcn_wave_barrier();
    asm volatile("s_wait_dscnt 0" ::: "memory");

    // logits: [16 q] x [32 keys] = 2 C tiles, K-dim = DH = 64 (2 wmma each)
    v8f c0 = {}, c1 = {};
#pragma unroll
    for (int kk = 0; kk < NDH; kk += 32) {
      v16h aq = afrag16x32(sQ + r16 * LQ_P + kk, hl);
      v16h b0 = bfrag32x16(sK + r16 * LK_P + kk + hl * 16);
      v16h b1 = bfrag32x16(sK + (16 + r16) * LK_P + kk + hl * 16);
      c0 = wmma_f16(aq, b0, c0);
      c1 = wmma_f16(aq, b1, c1);
    }

    // online softmax update (C layout: VGPR i -> rows i / i+8; lane -> col)
    int cg0 = kc + r16, cg1 = kc + 16 + r16;
#pragma unroll
    for (int i = 0; i < 8; ++i) {
      int qrow = q0 + i + 8 * hl;
      float s0 = (cg0 <= qrow) ? c0[i] * sc : -1e30f;
      float s1 = (cg1 <= qrow) ? c1[i] * sc : -1e30f;
      float mx = fmaxf(s0, s1);
      mx = fmaxf(mx, __shfl_xor(mx, 1, 32));
      mx = fmaxf(mx, __shfl_xor(mx, 2, 32));
      mx = fmaxf(mx, __shfl_xor(mx, 4, 32));
      mx = fmaxf(mx, __shfl_xor(mx, 8, 32));
      float mn = fmaxf(mrow[i], mx);
      float alpha = __expf(mrow[i] - mn);
      mrow[i] = mn;
      float p0 = __expf(s0 - mn);
      float p1 = __expf(s1 - mn);
      float ps = p0 + p1;
      ps += __shfl_xor(ps, 1, 32);
      ps += __shfl_xor(ps, 2, 32);
      ps += __shfl_xor(ps, 4, 32);
      ps += __shfl_xor(ps, 8, 32);
      lrow[i] = lrow[i] * alpha + ps;
#pragma unroll
      for (int t2 = 0; t2 < 4; ++t2) o[t2][i] *= alpha;
      int prow = i + 8 * hl;
      sP[prow * LP_P + r16]      = (_Float16)p0;
      sP[prow * LP_P + 16 + r16] = (_Float16)p1;
    }
    __builtin_amdgcn_wave_barrier();
    asm volatile("s_wait_dscnt 0" ::: "memory");

    // P(16x32) @ V(32x64): 4 n-tiles, shared A fragment
    v16h ap = afrag16x32(sP + r16 * LP_P, hl);
#pragma unroll
    for (int t2 = 0; t2 < 4; ++t2) {
      v16h bv = bfrag32x16(sV + (t2 * 16 + r16) * LV_P + hl * 16);
      o[t2] = wmma_f16(ap, bv, o[t2]);
    }
    __builtin_amdgcn_wave_barrier();
    asm volatile("s_wait_dscnt 0" ::: "memory");
  }

  // epilogue: divide by row sums, scatter into y[m, h*DH + dh]
#pragma unroll
  for (int i = 0; i < 8; ++i) {
    float inv = 1.0f / lrow[i];
    int row = q0 + i + 8 * hl;
    size_t m = (size_t)b * NS + row;
#pragma unroll
    for (int t2 = 0; t2 < 4; ++t2)
      Y[m * ND + h * NDH + t2 * 16 + r16] = (_Float16)(o[t2][i] * inv);
  }
}

// ---------------- Host orchestration ----------------------------------------

extern "C" void kernel_launch(void* const* d_in, const int* in_sizes, int n_in,
                              void* d_out, int out_size, void* d_ws, size_t ws_size,
                              hipStream_t stream) {
  (void)in_sizes; (void)n_in; (void)out_size; (void)ws_size;
  const float* states   = (const float*)d_in[0];
  const int*   actions  = (const int*)d_in[1];
  const int*   timesteps= (const int*)d_in[2];
  const float* W_s      = (const float*)d_in[3];
  const float* b_s      = (const float*)d_in[4];
  const float* A_emb    = (const float*)d_in[5];
  const float* pos_emb  = (const float*)d_in[6];
  const float* gpos_emb = (const float*)d_in[7];
  const float* ln1_g    = (const float*)d_in[8];
  const float* ln1_b    = (const float*)d_in[9];
  const float* Wq       = (const float*)d_in[10];
  const float* bq       = (const float*)d_in[11];
  const float* Wk       = (const float*)d_in[12];
  const float* bk       = (const float*)d_in[13];
  const float* Wv       = (const float*)d_in[14];
  const float* bv       = (const float*)d_in[15];
  const float* Wp       = (const float*)d_in[16];
  const float* bp       = (const float*)d_in[17];
  const float* ln2_g    = (const float*)d_in[18];
  const float* ln2_b    = (const float*)d_in[19];
  const float* W1       = (const float*)d_in[20];
  const float* b1       = (const float*)d_in[21];
  const float* W2       = (const float*)d_in[22];
  const float* b2       = (const float*)d_in[23];
  const float* lnf_g    = (const float*)d_in[24];
  const float* lnf_b    = (const float*)d_in[25];

  char* ws = (char*)d_ws;
  const size_t SZ_X = (size_t)MTOK * ND * sizeof(float);
  const size_t SZ_H = (size_t)MTOK * ND * sizeof(_Float16);
  const size_t SZ_G = (size_t)MTOK * NDFF * sizeof(_Float16);
  float*     x   = (float*)ws;                 ws += SZ_X;
  _Float16*  h16 = (_Float16*)ws;              ws += SZ_H;
  _Float16*  q16 = (_Float16*)ws;              ws += SZ_H;
  _Float16*  k16 = (_Float16*)ws;              ws += SZ_H;
  _Float16*  v16 = (_Float16*)ws;              ws += SZ_H;
  _Float16*  y16 = (_Float16*)ws;              ws += SZ_H;
  _Float16*  g16 = (_Float16*)ws;              ws += SZ_G;

  embed_kernel<<<NB * NT, 256, 0, stream>>>(states, actions, timesteps, W_s, b_s,
                                            A_emb, pos_emb, gpos_emb, x);

  dim3 gD(ND / BN, MTOK / BM);      // (6, 64)
  dim3 gF(NDFF / BN, MTOK / BM);    // (24, 64)

  for (int l = 0; l < NL; ++l) {
    const size_t wDD = (size_t)l * ND * ND;
    const size_t wDF = (size_t)l * ND * NDFF;
    ln_f16_kernel<<<MTOK, 256, 0, stream>>>(x, ln1_g + l * ND, ln1_b + l * ND, h16);
    gemm_kernel<0, 1><<<gD, 256, 0, stream>>>(h16, Wq + wDD, bq + (size_t)l * ND,
                                              nullptr, q16, MTOK, ND, ND);
    gemm_kernel<0, 1><<<gD, 256, 0, stream>>>(h16, Wk + wDD, bk + (size_t)l * ND,
                                              nullptr, k16, MTOK, ND, ND);
    gemm_kernel<0, 1><<<gD, 256, 0, stream>>>(h16, Wv + wDD, bv + (size_t)l * ND,
                                              nullptr, v16, MTOK, ND, ND);
    attn_kernel<<<(NB * NH * (NS / ANQ)) / AWAVES, 32 * AWAVES, 0, stream>>>(
        q16, k16, v16, y16);
    gemm_kernel<0, 2><<<gD, 256, 0, stream>>>(y16, Wp + wDD, bp + (size_t)l * ND,
                                              x, nullptr, MTOK, ND, ND);
    ln_f16_kernel<<<MTOK, 256, 0, stream>>>(x, ln2_g + l * ND, ln2_b + l * ND, h16);
    gemm_kernel<1, 0><<<gF, 256, 0, stream>>>(h16, W1 + wDF, b1 + (size_t)l * NDFF,
                                              nullptr, g16, MTOK, NDFF, ND);
    gemm_kernel<0, 2><<<gD, 256, 0, stream>>>(g16, W2 + wDF, b2 + (size_t)l * ND,
                                              x, nullptr, MTOK, ND, NDFF);
  }

  lnf_kernel<<<NB * NT, 256, 0, stream>>>(x, lnf_g, lnf_b, (float*)d_out);
}
